// Attention_21938692948404
// MI455X (gfx1250) — compile-verified, matrix operand order checked by reading, not run on hardware
//
#include <hip/hip_runtime.h>
#include <hip/hip_bf16.h>

typedef __attribute__((ext_vector_type(16))) __bf16 v16bf;
typedef __attribute__((ext_vector_type(8)))  float  v8f;

#define Bsz 32
#define Tsz 2048
#define Dsz 1024

// -------- workspace layout (bytes) --------
#define WS_WHHN   0                        // 32*1024*4  = 131072
#define WS_EM     131072                   // 65536*4    = 262144
#define WS_DENOM  393216                   // 32*4 (padded to 256)
#define WS_WYB    393472                   // 1024*1024*2 = 2 MiB (32B aligned)

#if __has_builtin(__builtin_amdgcn_tanhf)
#define fast_tanh(x) __builtin_amdgcn_tanhf(x)
#elif __has_builtin(__builtin_amdgcn_tanh_f32)
#define fast_tanh(x) __builtin_amdgcn_tanh_f32(x)
#else
#define fast_tanh(x) tanhf(x)
#endif

__device__ __forceinline__ unsigned short f2bf(float f) {
    unsigned int u = __float_as_uint(f);
    unsigned int r = u + 0x7FFFu + ((u >> 16) & 1u);   // round-to-nearest-even
    return (unsigned short)(r >> 16);
}

// ---------------------------------------------------------------------------
// Kernel 1a: whhn[b,e] = sum_d enc[b,d]*W_h[e,d]; zero d_out and denom.
// grid 128 x 256  (32768 threads = 32*1024 outputs)
// ---------------------------------------------------------------------------
__global__ void prep_whhn(const float* __restrict__ enc,
                          const float* __restrict__ W_h,
                          float* __restrict__ whhn,
                          float* __restrict__ out,
                          float* __restrict__ denom) {
    int gtid = blockIdx.x * 256 + threadIdx.x;
    int b = gtid >> 10;
    int e = gtid & 1023;
    const float4* er = reinterpret_cast<const float4*>(enc + (b << 10));
    const float4* wr = reinterpret_cast<const float4*>(W_h + (e << 10));
    float s = 0.f;
    #pragma unroll 4
    for (int d4 = 0; d4 < Dsz / 4; ++d4) {
        float4 a = er[d4], w = wr[d4];
        s += a.x * w.x + a.y * w.y + a.z * w.z + a.w * w.w;
    }
    whhn[gtid] = s;
    out[gtid] = 0.f;
    if (gtid < Bsz) denom[gtid] = 0.f;
}

// ---------------------------------------------------------------------------
// Kernel 1b: pack W_y (fp32, row-major [n][k]) into bf16 B-fragment layout:
//   p = ((nt*32 + kt)*32 + lane)*16 + i,  lane = N-col owner, i = K sequence
//   kl(h,i) = 8h + (i<8 ? i : 8+i),  h = lane>>4
// grid 4096 x 256
// ---------------------------------------------------------------------------
__global__ void pack_wy(const float* __restrict__ W_y,
                        unsigned short* __restrict__ wyb) {
    int p = blockIdx.x * 256 + threadIdx.x;
    int i    = p & 15;
    int lane = (p >> 4) & 31;
    int kt   = (p >> 9) & 31;
    int nt   = p >> 14;
    int n  = (nt << 4) + (lane & 15);
    int h  = lane >> 4;
    int kl = (h << 3) + ((i < 8) ? i : (8 + i));
    int k  = (kt << 5) + kl;
    wyb[p] = f2bf(W_y[n * Dsz + k]);
}

// ---------------------------------------------------------------------------
// Kernel 2: fused  tanh(X @ Wy^T + whhn) . w_a  -> sigmoid -> exp -> em, denom
// One block = 64 rows of X (all same batch b). 256 threads = 8 waves.
// wave w: wm = w&3 selects 16-row M subtile, nh = w>>2 selects N half.
// A tile staged fp32->bf16 in LDS (64x1024 bf16 = 128 KB) in fragment layout.
// Inner loop: v_wmma_f32_16x16x32_bf16, K accumulated over 32 k-tiles.
// Column reduction is deferred: 8 per-lane partial accumulators across the
// whole n loop, one butterfly at the end (v_tanh co-executes with WMMA).
// ---------------------------------------------------------------------------
__global__ void __launch_bounds__(256)
attn_logits(const float* __restrict__ X,        // [B*T, D] fp32
            const unsigned short* __restrict__ wyb,
            const float* __restrict__ whhn,     // [B, D]
            const float* __restrict__ w_a,      // [D]
            const float* __restrict__ mask,     // [B, T]
            float* __restrict__ em,             // [B*T] out: exp(sig)*mask
            float* __restrict__ denom) {        // [B]
    __shared__ unsigned short Albuf[64 * Dsz];  // 128 KB, bf16 A fragments
    __shared__ float lds_rowsum[64];

    const int tid  = threadIdx.x;
    const int lane = tid & 31;
    const int w    = tid >> 5;
    const int wm   = w & 3;
    const int nh   = w >> 2;

    const int row_base = blockIdx.x * 64;
    const int b  = row_base >> 11;          // T = 2048
    const int t0 = row_base & (Tsz - 1);

    if (tid < 64) lds_rowsum[tid] = 0.f;

    // ---- stage A tile: 64 rows x 1024 K, fp32 -> bf16, fragment layout ----
    // 16384 float4 groups, 64 per thread
    for (int it = tid; it < 16384; it += 256) {
        int gr = it >> 8;            // row in block 0..63
        int k  = (it & 255) << 2;    // k multiple of 4
        float4 v = *reinterpret_cast<const float4*>(
            X + (size_t)(row_base + gr) * Dsz + k);
        int kt = k >> 5, kl = k & 31;
        int h  = (kl & 15) >> 3;
        int i0 = (kl & 7) + ((kl >= 16) ? 8 : 0);
        int lane_s = (h << 4) + (gr & 15);
        int wm_s   = gr >> 4;
        int idx = (((wm_s << 5) + kt) << 9) + (lane_s << 4) + i0;
        unsigned int lo = (unsigned int)f2bf(v.x) | ((unsigned int)f2bf(v.y) << 16);
        unsigned int hi = (unsigned int)f2bf(v.z) | ((unsigned int)f2bf(v.w) << 16);
        *reinterpret_cast<uint2*>(&Albuf[idx]) = make_uint2(lo, hi);
    }
    __syncthreads();

    // ---- WMMA main loop ----
    const v16bf* Wy = reinterpret_cast<const v16bf*>(wyb);
    // per-lane column partials: colacc[i] = sum over this wave's n-tiles of
    // tanh(logit)*w_a for row (i + 8*(lane>>4)), col (lane&15) of each n-tile
    float colacc[8] = {0.f, 0.f, 0.f, 0.f, 0.f, 0.f, 0.f, 0.f};

    const int nt0 = nh << 5;                   // 32 n-tiles per wave
    for (int nt = nt0; nt < nt0 + 32; ++nt) {
        v8f acc = {0.f, 0.f, 0.f, 0.f, 0.f, 0.f, 0.f, 0.f};
        const v16bf* bp = Wy + ((size_t)nt << 10) + lane;  // (nt*32)*32 + lane
        __builtin_prefetch(bp, 0, 0);
        #pragma unroll 8
        for (int kt = 0; kt < 32; ++kt) {
            v16bf a = *reinterpret_cast<const v16bf*>(
                &Albuf[(((wm << 5) + kt) << 9) + (lane << 4)]);
            v16bf bb = bp[kt << 5];
            acc = __builtin_amdgcn_wmma_f32_16x16x32_bf16(
                false, a, false, bb, (short)0, acc, false, false);
        }
        // post-process: add whhn, tanh (V_TANH_F32), scale by w_a, accumulate
        int n = (nt << 4) + (lane & 15);
        float wh  = whhn[(b << 10) + n];
        float wav = w_a[n];
        #pragma unroll
        for (int i = 0; i < 8; ++i)
            colacc[i] += fast_tanh(acc[i] + wh) * wav;
    }

    // single butterfly per accumulator: reduce the 16 column-lanes per half
    if ((lane & 15) == 0) {
        // nothing: do reduction below for all lanes, deposit from lane 0/16
    }
    #pragma unroll
    for (int i = 0; i < 8; ++i) {
        float v = colacc[i];
        v += __shfl_xor(v, 1, 32);
        v += __shfl_xor(v, 2, 32);
        v += __shfl_xor(v, 4, 32);
        v += __shfl_xor(v, 8, 32);      // all 16 lanes of each half hold sum
        colacc[i] = v;
    }
    // combine the two N-halves across waves via LDS float atomics (ds_add_f32)
    if ((lane & 15) == 0) {
        int rbase = (wm << 4) + ((lane >> 4) << 3);
        #pragma unroll
        for (int i = 0; i < 8; ++i)
            atomicAdd(&lds_rowsum[rbase + i], colacc[i]);
    }
    __syncthreads();

    // ---- finalize: sigmoid, exp, mask; write em; atomic denom ----
    float ev = 0.f;
    if (tid < 64) {
        int r = tid;
        int t = t0 + r;
        float s  = lds_rowsum[r];
        float ad = 1.f / (1.f + __expf(-s));
        float e  = __expf(ad);
        float mv = mask[(b << 11) + t];
        ev = e * mv;
        em[(size_t)(b << 11) + t] = ev;
    }
    ev += __shfl_xor(ev, 1, 32);
    ev += __shfl_xor(ev, 2, 32);
    ev += __shfl_xor(ev, 4, 32);
    ev += __shfl_xor(ev, 8, 32);
    ev += __shfl_xor(ev, 16, 32);
    if (lane == 0 && w < 2) atomicAdd(&denom[b], ev);
}

// ---------------------------------------------------------------------------
// Kernel 3: out[b,d] = sum_t X[b,t,d] * em[b,t]/denom[b]
// grid (32, 16): each block handles one b, 128 t's, full D=1024 (4 per thread)
// ---------------------------------------------------------------------------
__global__ void __launch_bounds__(256)
pool(const float* __restrict__ X,
     const float* __restrict__ em,
     const float* __restrict__ denom,
     float* __restrict__ out) {
    int b  = blockIdx.x;
    int tc = blockIdx.y;
    int d0 = threadIdx.x << 2;
    float inv = 1.f / denom[b];
    float4 acc = make_float4(0.f, 0.f, 0.f, 0.f);
    #pragma unroll 4
    for (int tt = 0; tt < 128; ++tt) {
        int t = (tc << 7) + tt;
        float wgt = em[(size_t)b * Tsz + t] * inv;
        float4 x = *reinterpret_cast<const float4*>(
            X + ((size_t)b * Tsz + t) * Dsz + d0);
        acc.x += x.x * wgt; acc.y += x.y * wgt;
        acc.z += x.z * wgt; acc.w += x.w * wgt;
    }
    float* o = out + (size_t)b * Dsz + d0;
    atomicAdd(o + 0, acc.x);
    atomicAdd(o + 1, acc.y);
    atomicAdd(o + 2, acc.z);
    atomicAdd(o + 3, acc.w);
}

// ---------------------------------------------------------------------------
extern "C" void kernel_launch(void* const* d_in, const int* in_sizes, int n_in,
                              void* d_out, int out_size, void* d_ws, size_t ws_size,
                              hipStream_t stream) {
    const float* X    = (const float*)d_in[0];   // full_input [B,T,D]
    const float* enc  = (const float*)d_in[1];   // encoding   [B,D]
    const float* mask = (const float*)d_in[2];   // mask       [B,T]
    const float* W_h  = (const float*)d_in[3];   // [D,D]
    const float* W_y  = (const float*)d_in[4];   // [D,D]
    const float* w_a  = (const float*)d_in[5];   // [D]
    float* out = (float*)d_out;

    char* ws = (char*)d_ws;
    float*          whhn  = (float*)(ws + WS_WHHN);
    float*          em    = (float*)(ws + WS_EM);
    float*          denom = (float*)(ws + WS_DENOM);
    unsigned short* wyb   = (unsigned short*)(ws + WS_WYB);

    prep_whhn<<<128, 256, 0, stream>>>(enc, W_h, whhn, out, denom);
    pack_wy<<<4096, 256, 0, stream>>>(W_y, wyb);
    attn_logits<<<(Bsz * Tsz) / 64, 256, 0, stream>>>(X, wyb, whhn, w_a, mask,
                                                      em, denom);
    pool<<<dim3(Bsz, Tsz / 128), 256, 0, stream>>>(X, em, denom, out);
}
